// AttentionMechanism_8847632630316
// MI455X (gfx1250) — matrix operand order repaired and checked by reading, not compiled, not run on hardware
//
#include <hip/hip_runtime.h>
#include <math.h>

#define B_    32
#define KLEN  2000
#define KDIM  512
#define QDIM  512
#define ADIM  512
#define VDIM  512
#define CCH   10
#define CK    201

#define MTILE 32
#define KTILES ((KLEN + MTILE - 1) / MTILE)   // 63
#define CV_CHUNKS 16

typedef __attribute__((ext_vector_type(16))) __bf16 v16bf;
typedef __attribute__((ext_vector_type(8)))  float  v8f;

union BF16x16 {
    v16bf v;
    uint4 q[2];
    unsigned short s[16];
};

// ---------- fp32 -> bf16 hi/lo split helpers (RNE) ----------
__device__ __forceinline__ unsigned short f32_to_bf16(float x) {
    unsigned int u = __float_as_uint(x);
    unsigned int r = u + 0x7FFFu + ((u >> 16) & 1u);
    return (unsigned short)(r >> 16);
}
__device__ __forceinline__ float bf16_to_f32(unsigned short h) {
    return __uint_as_float((unsigned int)h << 16);
}

// ---------- kernel 1: split Wk into bf16 hi/lo (layout [a][d], row-major) ----------
__global__ __launch_bounds__(256) void prep_wk_kernel(
    const float* __restrict__ Wk,
    unsigned short* __restrict__ hi,
    unsigned short* __restrict__ lo)
{
    int i = blockIdx.x * 256 + threadIdx.x;
    if (i < ADIM * KDIM) {
        float x = Wk[i];
        unsigned short h = f32_to_bf16(x);
        hi[i] = h;
        lo[i] = f32_to_bf16(x - bf16_to_f32(h));
    }
}

// ---------- kernel 2: qb[b][a] = bk[a] + sum_d query[b][d]*Wq[a][d] ----------
__global__ __launch_bounds__(256) void prep_qb_kernel(
    const float* __restrict__ query,
    const float* __restrict__ Wq,
    const float* __restrict__ bk,
    float* __restrict__ qb)
{
    __shared__ float sq[QDIM];
    const int b = blockIdx.x;
    for (int i = threadIdx.x; i < QDIM; i += 256) sq[i] = query[(size_t)b * QDIM + i];
    __syncthreads();
    for (int a = threadIdx.x; a < ADIM; a += 256) {
        float acc = bk[a];
        const float4* wr = (const float4*)(Wq + (size_t)a * QDIM);
        const float4* qr = (const float4*)sq;
        #pragma unroll 4
        for (int d = 0; d < QDIM / 4; ++d) {
            float4 wv = wr[d], qv = qr[d];
            acc += wv.x * qv.x + wv.y * qv.y + wv.z * qv.z + wv.w * qv.w;
        }
        qb[(size_t)b * ADIM + a] = acc;
    }
}

// ---------- kernel 3: fused kproj (split-bf16 WMMA) + conv feat + tanh + v-dot -> e ----------
// LDS carve (bytes):
//   sAhi : 32 rows x 520 bf16 (8-elt row pad, de-conflicts banks)   33280
//   sAlo : same                                                     33280
//   sqb  : 512 f32                                                   2048
//   sv   : 512 f32                                                   2048
//   sconv: 32 x 10 f32 (conv features per row)                       1280
//   sered: 4(wn) x 32 f32                                             512
#define AROW 520
#define SM_AHI 0
#define SM_ALO 33280
#define SM_QB  66560
#define SM_V   68608
#define SM_CONV 70656
#define SM_ERED 71936
#define SMEM_BYTES 72448

__global__ __launch_bounds__(256) void kproj_score_kernel(
    const float* __restrict__ key,
    const float* __restrict__ aw_prev,
    const float* __restrict__ conv_w,
    const float* __restrict__ Wconv,
    const float* __restrict__ vvec,
    const unsigned short* __restrict__ wk_hi,
    const unsigned short* __restrict__ wk_lo,
    const float* __restrict__ qb,
    float* __restrict__ e_out)
{
    extern __shared__ char smem[];
    unsigned short* sAhi = (unsigned short*)(smem + SM_AHI);
    unsigned short* sAlo = (unsigned short*)(smem + SM_ALO);
    float* sqb   = (float*)(smem + SM_QB);
    float* sv    = (float*)(smem + SM_V);
    float* sconv = (float*)(smem + SM_CONV);
    float* sered = (float*)(smem + SM_ERED);

    const int tid = threadIdx.x;
    const int b   = blockIdx.x / KTILES;
    const int t   = blockIdx.x % KTILES;
    const int k0  = t * MTILE;

    // --- stage key tile (32 x 512 f32) -> bf16 hi/lo LDS, split once per element ---
    for (int i = tid; i < MTILE * KDIM; i += 256) {
        int row = i >> 9, col = i & (KDIM - 1);
        int gk = k0 + row;
        float x = (gk < KLEN) ? key[((size_t)b * KLEN + gk) * KDIM + col] : 0.0f;
        unsigned short h = f32_to_bf16(x);
        sAhi[row * AROW + col] = h;
        sAlo[row * AROW + col] = f32_to_bf16(x - bf16_to_f32(h));
    }
    // --- stage qb, v ---
    for (int i = tid; i < ADIM; i += 256) {
        sqb[i] = qb[(size_t)b * ADIM + i];
        sv[i]  = vvec[i];
    }
    // --- conv features: sconv[row][c] = sum_u aw_prev[b, k+u-100] * conv_w[c,u] ---
    for (int i = tid; i < MTILE * CCH; i += 256) {
        int row = i / CCH, c = i % CCH;
        int k = k0 + row;
        float acc = 0.0f;
        if (k < KLEN) {
            const float* wrow = conv_w + c * CK;
            const float* ap = aw_prev + (size_t)b * KLEN;
            for (int u = 0; u < CK; ++u) {
                int src = k + u - (CK / 2);
                if (src >= 0 && src < KLEN) acc += ap[src] * wrow[u];
            }
        }
        sconv[i] = acc;
    }
    if (tid < 128) sered[tid] = 0.0f;
    __syncthreads();

    // --- wave layout: 8 waves = 2 (M) x 4 (N); wave owns 16 rows x 128 cols ---
    const int lane = tid & 31;
    const int w  = tid >> 5;
    const int wm = w >> 2;        // 0..1  -> rows wm*16..+16
    const int wn = w & 3;         // 0..3  -> cols wn*128..+128
    const int li = lane & 15;
    const int hh = lane >> 4;

    v8f acc[8] = {};

    const unsigned short* ahp = sAhi + (wm * 16 + li) * AROW;
    const unsigned short* alp = sAlo + (wm * 16 + li) * AROW;

    for (int kk = 0; kk < KDIM; kk += 32) {
        // A fragment: lanes 0-15 hold K kk..kk+15,
        // lanes 16-31 hold K kk+16..kk+31 (matches B)
        BF16x16 ah, al;
        const int c0 = kk + hh * 16;
        ah.q[0] = *(const uint4*)(ahp + c0);
        ah.q[1] = *(const uint4*)(ahp + c0 + 8);
        al.q[0] = *(const uint4*)(alp + c0);
        al.q[1] = *(const uint4*)(alp + c0 + 8);

        #pragma unroll
        for (int j = 0; j < 8; ++j) {
            // B fragment: lane holds column a, 16 contiguous K values
            // (lanes 0-15: K kk..kk+15; lanes 16-31: K kk+16..kk+31)
            const int a = wn * 128 + j * 16 + li;
            const size_t boff = (size_t)a * KDIM + kk + hh * 16;
            BF16x16 bh, bl;
            bh.q[0] = *(const uint4*)(wk_hi + boff);
            bh.q[1] = *(const uint4*)(wk_hi + boff + 8);
            bl.q[0] = *(const uint4*)(wk_lo + boff);
            bl.q[1] = *(const uint4*)(wk_lo + boff + 8);
            // split-bf16 fp32 emulation: hi*hi + lo*hi + hi*lo
            acc[j] = __builtin_amdgcn_wmma_f32_16x16x32_bf16(false, ah.v, false, bh.v, (short)0, acc[j], false, false);
            acc[j] = __builtin_amdgcn_wmma_f32_16x16x32_bf16(false, al.v, false, bh.v, (short)0, acc[j], false, false);
            acc[j] = __builtin_amdgcn_wmma_f32_16x16x32_bf16(false, ah.v, false, bl.v, (short)0, acc[j], false, false);
        }
    }

    // --- epilogue: e contribution = v[a] * tanh(kproj + qb + convproj) ---
    // C layout: reg r -> M = hh*8 + r, N = li (within the wave's 16x16 fragment j)
    float ep[8] = {0.f, 0.f, 0.f, 0.f, 0.f, 0.f, 0.f, 0.f};
    #pragma unroll
    for (int j = 0; j < 8; ++j) {
        const int a = wn * 128 + j * 16 + li;
        const float va = sv[a];
        const float qa = sqb[a];
        float wc[CCH];
        #pragma unroll
        for (int c = 0; c < CCH; ++c) wc[c] = Wconv[a * CCH + c];   // L2-hot
        #pragma unroll
        for (int r = 0; r < 8; ++r) {
            const float* cr = sconv + (wm * 16 + hh * 8 + r) * CCH;
            float cf = 0.0f;
            #pragma unroll
            for (int c = 0; c < CCH; ++c) cf += cr[c] * wc[c];
            ep[r] += va * tanhf(acc[j][r] + qa + cf);
        }
    }
    // reduce across the 16 lanes sharing each row (xor masks stay inside lane-halves)
    #pragma unroll
    for (int r = 0; r < 8; ++r) {
        float s = ep[r];
        s += __shfl_xor(s, 1);
        s += __shfl_xor(s, 2);
        s += __shfl_xor(s, 4);
        s += __shfl_xor(s, 8);
        if (li == 0) sered[wn * 32 + wm * 16 + hh * 8 + r] = s;   // unique writer
    }
    __syncthreads();
    if (tid < MTILE) {
        int k = k0 + tid;
        if (k < KLEN)
            e_out[(size_t)b * KLEN + k] =
                sered[tid] + sered[32 + tid] + sered[64 + tid] + sered[96 + tid];
    }
}

// ---------- kernel 4: masked softmax over KLEN per batch ----------
__global__ __launch_bounds__(256) void softmax_kernel(
    const float* __restrict__ e,
    const int* __restrict__ mask,
    float* __restrict__ aw)
{
    __shared__ float red[256];
    const int b = blockIdx.x, tid = threadIdx.x;
    const float NEG = -3.402823466e38f;
    float vals[8];
    float m = NEG;
    #pragma unroll
    for (int i = 0; i < 8; ++i) {
        int k = tid + i * 256;
        float x = NEG;
        if (k < KLEN) {
            x = e[(size_t)b * KLEN + k];
            if (mask[(size_t)b * KLEN + k] == 0) x = NEG;
        }
        vals[i] = x;
        m = fmaxf(m, x);
    }
    red[tid] = m; __syncthreads();
    for (int s = 128; s > 0; s >>= 1) {
        if (tid < s) red[tid] = fmaxf(red[tid], red[tid + s]);
        __syncthreads();
    }
    m = red[0]; __syncthreads();

    float sum = 0.0f;
    #pragma unroll
    for (int i = 0; i < 8; ++i) {
        int k = tid + i * 256;
        float ex = (k < KLEN) ? __expf(vals[i] - m) : 0.0f;
        vals[i] = ex;
        sum += ex;
    }
    red[tid] = sum; __syncthreads();
    for (int s = 128; s > 0; s >>= 1) {
        if (tid < s) red[tid] += red[tid + s];
        __syncthreads();
    }
    const float inv = 1.0f / red[0];
    #pragma unroll
    for (int i = 0; i < 8; ++i) {
        int k = tid + i * 256;
        if (k < KLEN) aw[(size_t)b * KLEN + k] = vals[i] * inv;
    }
}

// ---------- kernel 5: cv partials (chunked over K for HBM concurrency) ----------
__global__ __launch_bounds__(512) void cv_partial_kernel(
    const float* __restrict__ aw,
    const float* __restrict__ value,
    float* __restrict__ part)
{
    const int b = blockIdx.x >> 4, ch = blockIdx.x & (CV_CHUNKS - 1);
    const int v = threadIdx.x;
    const int kbeg = ch * (KLEN / CV_CHUNKS);
    const int kend = kbeg + (KLEN / CV_CHUNKS);
    float acc = 0.0f;
    for (int k = kbeg; k < kend; ++k)
        acc += aw[(size_t)b * KLEN + k] * value[((size_t)b * KLEN + k) * VDIM + v];
    part[((size_t)b * CV_CHUNKS + ch) * VDIM + v] = acc;
}

// ---------- kernel 6: reduce partials -> cv ----------
__global__ __launch_bounds__(512) void cv_reduce_kernel(
    const float* __restrict__ part,
    float* __restrict__ cv)
{
    const int b = blockIdx.x, v = threadIdx.x;
    float s = 0.0f;
    #pragma unroll
    for (int ch = 0; ch < CV_CHUNKS; ++ch)
        s += part[((size_t)b * CV_CHUNKS + ch) * VDIM + v];
    cv[(size_t)b * VDIM + v] = s;
}

// ---------- workspace layout (bytes) ----------
#define WS_WKHI 0u          // 512KB  bf16 hi
#define WS_WKLO 524288u     // 512KB  bf16 lo
#define WS_QB   1048576u    // 64KB   qproj+bk
#define WS_E    1114112u    // 256KB  scores
#define WS_PART 1376256u    // 1MB    cv partials
// total ~2.4 MB

extern "C" void kernel_launch(void* const* d_in, const int* in_sizes, int n_in,
                              void* d_out, int out_size, void* d_ws, size_t ws_size,
                              hipStream_t stream) {
    const float* key     = (const float*)d_in[0];
    const float* value   = (const float*)d_in[1];
    const float* query   = (const float*)d_in[2];
    const float* aw_prev = (const float*)d_in[3];
    const int*   mask    = (const int*)d_in[4];
    const float* Wk      = (const float*)d_in[5];
    const float* bk      = (const float*)d_in[6];
    const float* Wq      = (const float*)d_in[7];
    const float* conv_w  = (const float*)d_in[8];
    const float* Wconv   = (const float*)d_in[9];
    const float* v       = (const float*)d_in[10];

    float* out    = (float*)d_out;
    float* cv_out = out;                          // [B,1,VDIM]
    float* aw_out = out + (size_t)B_ * VDIM;      // [B,1,1,KLEN]

    char* ws = (char*)d_ws;
    unsigned short* wk_hi = (unsigned short*)(ws + WS_WKHI);
    unsigned short* wk_lo = (unsigned short*)(ws + WS_WKLO);
    float* qb   = (float*)(ws + WS_QB);
    float* e    = (float*)(ws + WS_E);
    float* part = (float*)(ws + WS_PART);

    prep_wk_kernel<<<(ADIM * KDIM + 255) / 256, 256, 0, stream>>>(Wk, wk_hi, wk_lo);
    prep_qb_kernel<<<B_, 256, 0, stream>>>(query, Wq, bk, qb);
    kproj_score_kernel<<<B_ * KTILES, 256, SMEM_BYTES, stream>>>(
        key, aw_prev, conv_w, Wconv, v, wk_hi, wk_lo, qb, e);
    softmax_kernel<<<B_, 256, 0, stream>>>(e, mask, aw_out);
    cv_partial_kernel<<<B_ * CV_CHUNKS, 512, 0, stream>>>(aw_out, value, part);
    cv_reduce_kernel<<<B_, 512, 0, stream>>>(part, cv_out);
}
